// CVM_VIGOR_1322849927576
// MI455X (gfx1250) — matrix operand order repaired and checked by reading, not compiled, not run on hardware
//
#include <hip/hip_runtime.h>
#include <hip/hip_bf16.h>
#include <math.h>

// ---------------- problem constants ----------------
#define EMBED     128
#define HEADS     8
#define HEAD_DIM  16
#define LEVELS    5
#define POINTS    4
#define QDIM      256
#define NQ        (QDIM * QDIM)          // 65536
#define LTOT      87296                  // 65536+16384+4096+1024+256
#define NOFF      (HEADS * LEVELS * POINTS * 2)  // 320
#define NATT      (HEADS * LEVELS * POINTS)      // 160

typedef __attribute__((ext_vector_type(16))) __bf16 bf16x16;
typedef __attribute__((ext_vector_type(8)))  float  floatx8;

__device__ __forceinline__ unsigned short f2bf(float f) {
    unsigned int u = __float_as_uint(f);
    unsigned int r = u + 0x7FFFu + ((u >> 16) & 1u);
    return (unsigned short)(r >> 16);
}

// ---------------- stage 1: 1x1 conv projection (per level) ----------------
// x[(baseRow+pos)*128 + o] = b[o] + sum_c w[o*C+c] * f[c*HW + pos]
__global__ void proj_kernel(const float* __restrict__ f, const float* __restrict__ w,
                            const float* __restrict__ b, float* __restrict__ x,
                            int baseRow, int HW, int C) {
    int tid = blockIdx.x * blockDim.x + threadIdx.x;
    if (tid >= HW * EMBED) return;
    int o   = tid & 127;
    int pos = tid >> 7;
    float s = b[o];
    const float* wr = w + (size_t)o * C;
    for (int c = 0; c < C; ++c) s = fmaf(wr[c], f[(size_t)c * HW + pos], s);
    x[(((size_t)(baseRow + pos)) << 7) + o] = s;
}

// ---------------- stage 2: GroupNorm stats (32 groups per level) ----------------
__global__ void gn_stats_kernel(const float* __restrict__ x, float* __restrict__ stats,
                                int baseRow, int HW, int statBase) {
    int g = blockIdx.x;               // 0..31
    __shared__ float ssum[256];
    __shared__ float ssq[256];
    float s = 0.f, q2 = 0.f;
    int n = HW * 4;
    for (int i = threadIdx.x; i < n; i += 256) {
        int pos = i >> 2;
        int o   = (g << 2) + (i & 3);
        float v = x[(((size_t)(baseRow + pos)) << 7) + o];
        s += v; q2 += v * v;
    }
    ssum[threadIdx.x] = s; ssq[threadIdx.x] = q2;
    __syncthreads();
    for (int off = 128; off > 0; off >>= 1) {
        if (threadIdx.x < off) {
            ssum[threadIdx.x] += ssum[threadIdx.x + off];
            ssq[threadIdx.x]  += ssq[threadIdx.x + off];
        }
        __syncthreads();
    }
    if (threadIdx.x == 0) {
        float mu  = ssum[0] / (float)n;
        float var = ssq[0] / (float)n - mu * mu;
        stats[(size_t)(statBase + g) * 2]     = mu;
        stats[(size_t)(statBase + g) * 2 + 1] = rsqrtf(var + 1e-5f);
    }
}

// ---------------- stage 3: GroupNorm apply (in place) + bf16 copy ----------------
__global__ void gn_apply_kernel(float* __restrict__ x, unsigned short* __restrict__ xbf,
                                const float* __restrict__ stats,
                                const float* __restrict__ gw, const float* __restrict__ gb,
                                int baseRow, int HW, int statBase) {
    int tid = blockIdx.x * blockDim.x + threadIdx.x;
    if (tid >= HW * EMBED) return;
    int o   = tid & 127;
    int pos = tid >> 7;
    int g   = o >> 2;
    float mu   = stats[(size_t)(statBase + g) * 2];
    float rstd = stats[(size_t)(statBase + g) * 2 + 1];
    size_t idx = (((size_t)(baseRow + pos)) << 7) + o;
    float v = (x[idx] - mu) * rstd * gw[o] + gb[o];
    x[idx]   = v;
    xbf[idx] = f2bf(v);
}

// ---------------- f32 -> bf16 elementwise ----------------
__global__ void cvt_bf16_kernel(const float* __restrict__ in, unsigned short* __restrict__ out,
                                size_t n) {
    size_t tid = (size_t)blockIdx.x * blockDim.x + threadIdx.x;
    if (tid < n) out[tid] = f2bf(in[tid]);
}

// ---------------- pack f32 weight [K][N] into WMMA B fragments (bf16) ----------
// layout: Bp[((tn*ksteps + ks)*32 + lane)*16 + j]
//   lane<16 : N = tn*16+lane,    K = ks*32 + j        (j = 2v+h, VGPR v half h)
//   lane>=16: N = tn*16+lane-16, K = ks*32 + 16 + j
__global__ void pack_b_kernel(const float* __restrict__ B, unsigned short* __restrict__ Bp,
                              int K, int N) {
    int tid = blockIdx.x * blockDim.x + threadIdx.x;
    int ksteps = K >> 5;
    int total = (N >> 4) * ksteps * 32;
    if (tid >= total) return;
    int lane = tid & 31;
    int ks   = (tid >> 5) % ksteps;
    int tn   = (tid >> 5) / ksteps;
    int n     = (tn << 4) + (lane & 15);
    int kbase = (ks << 5) + ((lane >> 4) << 4);
    unsigned short* dst = Bp + (size_t)tid * 16;
    #pragma unroll
    for (int j = 0; j < 16; ++j)
        dst[j] = f2bf(B[(size_t)(kbase + j) * N + n]);
}

// ---------------- WMMA bf16 GEMM: C[M,N] = A[M,K]*B[K,N] + bias (+residual) ----
// one 16x16 tile per wave, 8 waves per block; K stepped by 32
__global__ void gemm_bf16_kernel(const unsigned short* __restrict__ A,
                                 const unsigned short* __restrict__ Bp,
                                 const float* __restrict__ bias,
                                 const float* __restrict__ residual,
                                 float* __restrict__ C,
                                 int M, int N, int K) {
    int lane = threadIdx.x & 31;
    int wave = threadIdx.x >> 5;
    int tileM = (blockIdx.x * 8 + wave) * 16;
    int tileN = blockIdx.y * 16;
    if (tileM >= M) return;
    int ksteps = K >> 5;

    floatx8 acc = {0.f, 0.f, 0.f, 0.f, 0.f, 0.f, 0.f, 0.f};

    // A fragment source: row = tileM + (lane&15); two 16B chunks per k-step
    const unsigned short* arow = A + (size_t)(tileM + (lane & 15)) * K + ((lane >> 4) << 4);
    // B fragment source: one contiguous 32B chunk per (ks, lane)
    const unsigned short* bbase = Bp + (((size_t)blockIdx.y * ksteps) * 32 + lane) * 16;

    for (int ks = 0; ks < ksteps; ++ks) {
        union { uint4 u[2]; bf16x16 v; } af, bf;
        af.u[0] = *reinterpret_cast<const uint4*>(arow + ks * 32);
        af.u[1] = *reinterpret_cast<const uint4*>(arow + ks * 32 + 8);
        const uint4* bp = reinterpret_cast<const uint4*>(bbase + (size_t)ks * 32 * 16);
        bf.u[0] = bp[0];
        bf.u[1] = bp[1];
        acc = __builtin_amdgcn_wmma_f32_16x16x32_bf16(
            false, af.v, false, bf.v, (short)0, acc, false, false);
    }

    int n     = tileN + (lane & 15);
    int mbase = tileM + ((lane >> 4) << 3);
    float bn = bias[n];
    #pragma unroll
    for (int r = 0; r < 8; ++r) {
        size_t idx = (size_t)(mbase + r) * N + n;
        float v = acc[r] + bn;
        if (residual) v += residual[idx];
        C[idx] = v;
    }
}

// ---------------- softmax over 20 (per query, per head) ----------------
__global__ void softmax20_kernel(float* __restrict__ attn, int total) {
    int t = blockIdx.x * blockDim.x + threadIdx.x;
    if (t >= total) return;
    float* a = attn + (size_t)t * 20;
    float m = a[0];
    #pragma unroll
    for (int i = 1; i < 20; ++i) m = fmaxf(m, a[i]);
    float s = 0.f;
    #pragma unroll
    for (int i = 0; i < 20; ++i) { float e = __expf(a[i] - m); a[i] = e; s += e; }
    float inv = 1.f / s;
    #pragma unroll
    for (int i = 0; i < 20; ++i) a[i] *= inv;
}

// ---------------- bilinear multi-scale sampling -> agg (NQ,128) ----------------
__device__ __forceinline__ float tap(const float* __restrict__ v, int base, int W, int H,
                                     int xi, int yi, int c) {
    if (xi < 0 || xi >= W || yi < 0 || yi >= H) return 0.f;
    return v[(((size_t)(base + yi * W + xi)) << 7) + c];
}

__global__ void sample_kernel(const float* __restrict__ value,
                              const float* __restrict__ offs,
                              const float* __restrict__ attn,
                              float* __restrict__ agg) {
    int tid = blockIdx.x * blockDim.x + threadIdx.x;   // over NQ*128
    if (tid >= NQ * EMBED) return;
    int c = tid & 127;
    int q = tid >> 7;
    int h = c >> 4;
    float refx = ((float)(q & 255) + 0.5f) * (1.0f / 256.0f);
    float refy = ((float)(q >> 8)  + 0.5f) * (1.0f / 256.0f);
    const float* offq = offs + (size_t)q * NOFF + h * (LEVELS * POINTS * 2);
    const float* attq = attn + (size_t)q * NATT + h * (LEVELS * POINTS);

    const int Hs[5] = {256, 128, 64, 32, 16};
    const int st[5] = {0, 65536, 81920, 86016, 87040};

    float acc = 0.f;
    #pragma unroll
    for (int lvl = 0; lvl < LEVELS; ++lvl) {
        int H = Hs[lvl], W = Hs[lvl];
        int base = st[lvl];
        #pragma unroll
        for (int p = 0; p < POINTS; ++p) {
            float ox = offq[lvl * 8 + p * 2];
            float oy = offq[lvl * 8 + p * 2 + 1];
            float a  = attq[lvl * 4 + p];
            // loc = ref + off/[W,H];  x = loc.x*W - 0.5 = refx*W + ox - 0.5
            float x = refx * (float)W + ox - 0.5f;
            float y = refy * (float)H + oy - 0.5f;
            float x0f = floorf(x), y0f = floorf(y);
            float wx = x - x0f, wy = y - y0f;
            int x0 = (int)x0f, y0 = (int)y0f;
            int x1 = x0 + 1,   y1 = y0 + 1;
            float s = tap(value, base, W, H, x0, y0, c) * ((1.f - wx) * (1.f - wy))
                    + tap(value, base, W, H, x1, y0, c) * (wx * (1.f - wy))
                    + tap(value, base, W, H, x0, y1, c) * ((1.f - wx) * wy)
                    + tap(value, base, W, H, x1, y1, c) * (wx * wy);
            acc = fmaf(a, s, acc);
        }
    }
    agg[tid] = acc;
}

// ---------------- host side ----------------
static inline size_t alignUp(size_t x, size_t a) { return (x + a - 1) & ~(a - 1); }

extern "C" void kernel_launch(void* const* d_in, const int* in_sizes, int n_in,
                              void* d_out, int out_size, void* d_ws, size_t ws_size,
                              hipStream_t stream) {
    (void)in_sizes; (void)n_in; (void)out_size; (void)ws_size;

    // ---- input pointers (setup_inputs dict order, params flattened) ----
    const float* f[5]      = {(const float*)d_in[0], (const float*)d_in[1], (const float*)d_in[2],
                              (const float*)d_in[3], (const float*)d_in[4]};
    const float* proj_w[5] = {(const float*)d_in[5],  (const float*)d_in[9],  (const float*)d_in[13],
                              (const float*)d_in[17], (const float*)d_in[21]};
    const float* proj_b[5] = {(const float*)d_in[6],  (const float*)d_in[10], (const float*)d_in[14],
                              (const float*)d_in[18], (const float*)d_in[22]};
    const float* gn_w[5]   = {(const float*)d_in[7],  (const float*)d_in[11], (const float*)d_in[15],
                              (const float*)d_in[19], (const float*)d_in[23]};
    const float* gn_b[5]   = {(const float*)d_in[8],  (const float*)d_in[12], (const float*)d_in[16],
                              (const float*)d_in[20], (const float*)d_in[24]};
    const float* query  = (const float*)d_in[25];
    const float* W_off  = (const float*)d_in[26];
    const float* b_off  = (const float*)d_in[27];
    const float* W_attn = (const float*)d_in[28];
    const float* b_attn = (const float*)d_in[29];
    const float* W_val  = (const float*)d_in[30];
    const float* b_val  = (const float*)d_in[31];
    const float* W_out  = (const float*)d_in[32];
    const float* b_out  = (const float*)d_in[33];
    float* out = (float*)d_out;

    const int HWs[5] = {65536, 16384, 4096, 1024, 256};
    const int Cs[5]  = {16, 24, 40, 112, 320};
    const int baseRow[5] = {0, 65536, 81920, 86016, 87040};

    // ---- workspace carve-up ----
    char* ws = (char*)d_ws;
    size_t off = 0;
    auto carve = [&](size_t bytes) { size_t o = off; off = alignUp(off + bytes, 256); return o; };

    float*          xproj  = (float*)(ws + carve((size_t)LTOT * EMBED * 4));  // also holds GN'd vpre
    unsigned short* vprebf = (unsigned short*)(ws + carve((size_t)LTOT * EMBED * 2));
    float*          value  = (float*)(ws + carve((size_t)LTOT * EMBED * 4));
    unsigned short* qbf    = (unsigned short*)(ws + carve((size_t)NQ * EMBED * 2));
    float*          offs   = (float*)(ws + carve((size_t)NQ * NOFF * 4));
    float*          attn   = (float*)(ws + carve((size_t)NQ * NATT * 4));
    float*          agg    = (float*)(ws + carve((size_t)NQ * EMBED * 4));
    unsigned short* aggbf  = (unsigned short*)(ws + carve((size_t)NQ * EMBED * 2));
    float*          stats  = (float*)(ws + carve((size_t)LEVELS * 32 * 2 * 4));
    unsigned short* pWval  = (unsigned short*)(ws + carve((size_t)EMBED * EMBED * 2));
    unsigned short* pWoff  = (unsigned short*)(ws + carve((size_t)EMBED * NOFF * 2));
    unsigned short* pWattn = (unsigned short*)(ws + carve((size_t)EMBED * NATT * 2));
    unsigned short* pWout  = (unsigned short*)(ws + carve((size_t)EMBED * EMBED * 2));

    // ---- stage 1-3: projection + GroupNorm per level ----
    for (int lvl = 0; lvl < LEVELS; ++lvl) {
        int HW = HWs[lvl];
        int nthr = HW * EMBED;
        proj_kernel<<<(nthr + 255) / 256, 256, 0, stream>>>(
            f[lvl], proj_w[lvl], proj_b[lvl], xproj, baseRow[lvl], HW, Cs[lvl]);
        gn_stats_kernel<<<32, 256, 0, stream>>>(xproj, stats, baseRow[lvl], HW, lvl * 32);
        gn_apply_kernel<<<(nthr + 255) / 256, 256, 0, stream>>>(
            xproj, vprebf, stats, gn_w[lvl], gn_b[lvl], baseRow[lvl], HW, lvl * 32);
    }

    // ---- stage 4: value = vpre @ W_val + b_val (WMMA bf16) ----
    {
        int total = (EMBED / 16) * (EMBED / 32) * 32;
        pack_b_kernel<<<(total + 255) / 256, 256, 0, stream>>>(W_val, pWval, EMBED, EMBED);
        dim3 grid(LTOT / 128, EMBED / 16);
        gemm_bf16_kernel<<<grid, 256, 0, stream>>>(vprebf, pWval, b_val, nullptr, value,
                                                   LTOT, EMBED, EMBED);
    }

    // ---- stage 5: q projections (offsets + attention logits) ----
    cvt_bf16_kernel<<<((size_t)NQ * EMBED + 255) / 256, 256, 0, stream>>>(query, qbf,
                                                                          (size_t)NQ * EMBED);
    {
        int total = (NOFF / 16) * (EMBED / 32) * 32;
        pack_b_kernel<<<(total + 255) / 256, 256, 0, stream>>>(W_off, pWoff, EMBED, NOFF);
        dim3 grid(NQ / 128, NOFF / 16);
        gemm_bf16_kernel<<<grid, 256, 0, stream>>>(qbf, pWoff, b_off, nullptr, offs,
                                                   NQ, NOFF, EMBED);
    }
    {
        int total = (NATT / 16) * (EMBED / 32) * 32;
        pack_b_kernel<<<(total + 255) / 256, 256, 0, stream>>>(W_attn, pWattn, EMBED, NATT);
        dim3 grid(NQ / 128, NATT / 16);
        gemm_bf16_kernel<<<grid, 256, 0, stream>>>(qbf, pWattn, b_attn, nullptr, attn,
                                                   NQ, NATT, EMBED);
    }
    softmax20_kernel<<<(NQ * HEADS + 255) / 256, 256, 0, stream>>>(attn, NQ * HEADS);

    // ---- stage 6: deformable sampling -> agg ----
    sample_kernel<<<((size_t)NQ * EMBED + 255) / 256, 256, 0, stream>>>(value, offs, attn, agg);

    // ---- stage 7: out = agg @ W_out + b_out + q (WMMA bf16 + f32 residual) ----
    cvt_bf16_kernel<<<((size_t)NQ * EMBED + 255) / 256, 256, 0, stream>>>(agg, aggbf,
                                                                          (size_t)NQ * EMBED);
    {
        int total = (EMBED / 16) * (EMBED / 32) * 32;
        pack_b_kernel<<<(total + 255) / 256, 256, 0, stream>>>(W_out, pWout, EMBED, EMBED);
        dim3 grid(NQ / 128, EMBED / 16);
        gemm_bf16_kernel<<<grid, 256, 0, stream>>>(aggbf, pWout, b_out, query, out,
                                                   NQ, EMBED, EMBED);
    }
}
